// RnnModule_27015344292064
// MI455X (gfx1250) — matrix operand order repaired and checked by reading, not compiled, not run on hardware
//
#include <hip/hip_runtime.h>
#include <math.h>

// Problem constants (reference: B=128, T=1024, H=512)
#define B_ 128
#define T_ 1024
#define H_ 512

typedef __attribute__((ext_vector_type(16))) __bf16       v16bf;
typedef __attribute__((ext_vector_type(2)))  __bf16       v2bf;
typedef __attribute__((ext_vector_type(8)))  float        v8f;
typedef __attribute__((ext_vector_type(8)))  unsigned int v8u;
typedef __attribute__((ext_vector_type(4)))  unsigned int u32x4;
typedef __attribute__((ext_vector_type(8)))  int          i32x8;
typedef __attribute__((ext_vector_type(4)))  int          i32x4;

#if defined(__has_builtin)
#if __has_builtin(__builtin_amdgcn_tensor_load_to_lds)
#define HAVE_TDM 1
#endif
#if __has_builtin(__builtin_amdgcn_tanhf)
#define FAST_TANH(x) __builtin_amdgcn_tanhf(x)
#elif __has_builtin(__builtin_amdgcn_tanh_f32)
#define FAST_TANH(x) __builtin_amdgcn_tanh_f32(x)
#endif
#endif
#ifndef HAVE_TDM
#define HAVE_TDM 0
#endif
#ifndef FAST_TANH
__device__ __forceinline__ float __fast_tanh_asm(float x) {
  float r;
  asm("v_tanh_f32 %0, %1" : "=v"(r) : "v"(x));   // CDNA5 native TANH trans op
  return r;
}
#define FAST_TANH(x) __fast_tanh_asm(x)
#endif

#define WMMA_BF16(a, b, c) \
  __builtin_amdgcn_wmma_f32_16x16x32_bf16(false, (a), false, (b), (short)0, (c), false, false)

// ---- bf16 helpers ---------------------------------------------------------
// Plain C __bf16 conversion (RNE): lets the backend select the hardware
// v_cvt_pk_bf16_f32 pattern when available instead of a manual bit sequence.
__device__ __forceinline__ unsigned pack_bf16(float lo, float hi) {
  v2bf p;
  p[0] = (__bf16)lo;
  p[1] = (__bf16)hi;
  return __builtin_bit_cast(unsigned, p);
}
__device__ __forceinline__ unsigned short bf16_1(float x) {
  __bf16 h = (__bf16)x;
  return __builtin_bit_cast(unsigned short, h);
}

// A tile (16x32 bf16) from an f32 row: p -> A[row][k0+koff], koff = 0|8.
// v0..v3 hold K=koff..koff+7, v4..v7 hold K=koff+16..koff+23 (ISA 7.12.2).
__device__ __forceinline__ v16bf load_a_f32(const float* __restrict__ p) {
  const float4* q = (const float4*)p;
  const float4* r = (const float4*)(p + 16);
  float4 x0 = q[0], x1 = q[1], y0 = r[0], y1 = r[1];
  v8u u;
  u[0] = pack_bf16(x0.x, x0.y); u[1] = pack_bf16(x0.z, x0.w);
  u[2] = pack_bf16(x1.x, x1.y); u[3] = pack_bf16(x1.z, x1.w);
  u[4] = pack_bf16(y0.x, y0.y); u[5] = pack_bf16(y0.z, y0.w);
  u[6] = pack_bf16(y1.x, y1.y); u[7] = pack_bf16(y1.z, y1.w);
  return __builtin_bit_cast(v16bf, u);
}

// A tile from a pre-packed bf16 row-major state buffer: two b128 loads only.
__device__ __forceinline__ v16bf load_a_bf(const unsigned short* __restrict__ p) {
  uint4 a = *(const uint4*)p;
  uint4 b = *(const uint4*)(p + 16);
  v8u u = {a.x, a.y, a.z, a.w, b.x, b.y, b.z, b.w};
  return __builtin_bit_cast(v16bf, u);
}

// B tile (32x16 bf16, pre-swizzled, 256 dwords laid out [lane][vgpr0..7]).
__device__ __forceinline__ v16bf load_b_sw(const unsigned* __restrict__ tile, int lane) {
  const uint4* p = (const uint4*)(tile + lane * 8);
  uint4 a = p[0], b = p[1];
  v8u u = {a.x, a.y, a.z, a.w, b.x, b.y, b.z, b.w};
  return __builtin_bit_cast(v16bf, u);
}

// ---------------------------------------------------------------------------
// One-time: swizzle [H,H] f32 weights into bf16 WMMA-B tile layout.
// ---------------------------------------------------------------------------
__global__ __launch_bounds__(256) void swizzle_w_bf16(const float* __restrict__ W,
                                                      unsigned* __restrict__ dst) {
  const int wave = blockIdx.x * 8 + (threadIdx.x >> 5);
  const int lane = threadIdx.x & 31;
  if (wave >= (H_ / 32) * (H_ / 16)) return;
  const int kt = wave >> 5;
  const int nt = wave & 31;
  const int n = nt * 16 + (lane & 15);
  const int kbase = kt * 32 + ((lane >> 4) << 3);
  unsigned* out = dst + (size_t)wave * 256 + lane * 8;
#pragma unroll
  for (int j = 0; j < 8; ++j) {
    float lo = W[(size_t)(kbase + 2 * j + (j >= 4 ? 8 : 0))     * H_ + n];
    float hi = W[(size_t)(kbase + 2 * j + (j >= 4 ? 8 : 0) + 1) * H_ + n];
    out[j] = pack_bf16(lo, hi);
  }
}

// ---------------------------------------------------------------------------
// Phase 1: Xp[t,b,:] = x[b,t,:] @ Wx + bias  (into d_out, [T,B,H], f32).
// One-deep software pipeline: loads of slab kk+1 are issued before the four
// WMMAs of slab kk, so memory latency hides under the matrix chain.
// ---------------------------------------------------------------------------
__global__ __launch_bounds__(256) void gemm_x_wx(const float* __restrict__ X,
                                                 const unsigned* __restrict__ Wsw,
                                                 const float* __restrict__ bias,
                                                 float* __restrict__ out) {
  const int lane = threadIdx.x & 31;
  const int wave = blockIdx.x * 8 + (threadIdx.x >> 5);
  const int rt = wave >> 3;
  const int nb = wave & 7;
  const int r0 = rt << 4;
  const int t  = r0 >> 7;
  const int b0 = r0 & 127;
  const int n0 = nb << 6;

  const int mA   = lane & 15;
  const int koff = (lane >> 4) << 3;
  const float* arow = X + ((size_t)(b0 + mA) * T_ + t) * H_ + koff;

  const unsigned* bt = Wsw + (size_t)(n0 >> 4) * 256;
  v8f acc0 = {}, acc1 = {}, acc2 = {}, acc3 = {};

  v16bf a   = load_a_f32(arow);
  v16bf b0v = load_b_sw(bt,       lane);
  v16bf b1v = load_b_sw(bt + 256, lane);
  v16bf b2v = load_b_sw(bt + 512, lane);
  v16bf b3v = load_b_sw(bt + 768, lane);

#pragma unroll
  for (int kk = 0; kk < 16; ++kk) {
    v16bf ca = a, c0 = b0v, c1 = b1v, c2 = b2v, c3 = b3v;
    if (kk < 15) {
      bt += 32 * 256;
      a   = load_a_f32(arow + (kk + 1) * 32);
      b0v = load_b_sw(bt,       lane);
      b1v = load_b_sw(bt + 256, lane);
      b2v = load_b_sw(bt + 512, lane);
      b3v = load_b_sw(bt + 768, lane);
      __builtin_prefetch(bt + 32 * 256 + lane * 8, 0, 1);
    }
    acc0 = WMMA_BF16(ca, c0, acc0);
    acc1 = WMMA_BF16(ca, c1, acc1);
    acc2 = WMMA_BF16(ca, c2, acc2);
    acc3 = WMMA_BF16(ca, c3, acc3);
  }

  const int nc = lane & 15;
  const int mc = (lane >> 4) << 3;
#pragma unroll
  for (int i = 0; i < 8; ++i) {
    size_t ob = ((size_t)t * B_ + (b0 + mc + i)) * H_ + n0 + nc;
    out[ob]      = acc0[i] + bias[n0 + nc];
    out[ob + 16] = acc1[i] + bias[n0 + 16 + nc];
    out[ob + 32] = acc2[i] + bias[n0 + 32 + nc];
    out[ob + 48] = acc3[i] + bias[n0 + 48 + nc];
  }
}

// ---------------------------------------------------------------------------
// Phase 2 (per step): out_t = tanh(Xp_t + state_{t-1} @ Wh), in place on
// d_out[t]; emits a bf16 row-major copy of the new state for step t+1.
// Block = one 64-wide N slab; wave = one 16-row M tile. The block's 64 KB
// slab of pre-swizzled Wh is DMA'd into LDS by the Tensor Data Mover; the
// first A (state) load overlaps the DMA.
// ---------------------------------------------------------------------------
__global__ __launch_bounds__(256) void rnn_step(float* __restrict__ xp,
                                                const unsigned short* __restrict__ sb_prev,
                                                unsigned short* __restrict__ sb_next,
                                                const unsigned* __restrict__ Wsw,
                                                int first) {
  const int lane = threadIdx.x & 31;
  const int wv   = threadIdx.x >> 5;        // 0..7 -> M tile
  const int m0 = wv << 4;
  const int n0 = blockIdx.x << 6;

#if HAVE_TDM
  __shared__ unsigned lds_wh[16 * 1024];    // 64 KB: [kk][tile j][lane][8 dwords]
  if (wv == 0) {
    // D# group 0: count=1, lds_addr, global_addr[56:0], type=2 (ISA §8.3).
    unsigned long long ga =
        (unsigned long long)(const void*)Wsw + (unsigned long long)blockIdx.x * 4096ull;
    unsigned lds_off = (unsigned)(unsigned long long)(const void*)&lds_wh[0];
    u32x4 g0;
    g0[0] = 1u;                                        // count=1 (valid), user mode
    g0[1] = lds_off;                                   // lds_addr
    g0[2] = (unsigned)ga;                              // global_addr[31:0]
    g0[3] = (unsigned)((ga >> 32) & 0x01FFFFFFu) | (2u << 30);  // addr[56:32]|type=2
    // D# group 1: data_size=4B; tensor 1024x16 dwords; tile 1024x16;
    // dim0 stride = 8192 dwords (= 32 KB between K-slabs) (ISA §8.4).
    i32x8 g1;
    g1[0] = (int)(2u << 16);                           // data_size=2 (4 bytes)
    g1[1] = (int)((1024u & 0xFFFFu) << 16);            // tensor_dim0[15:0] in [31:16]
    g1[2] = (int)(16u << 16);                          // tensor_dim0 hi=0; tensor_dim1=16
    g1[3] = (int)(1024u << 16);                        // tensor_dim1 hi=0; tile_dim0=1024
    g1[4] = 16;                                        // tile_dim1=16; tile_dim2=0
    g1[5] = 8192;                                      // tensor_dim0_stride lo
    g1[6] = 0;
    g1[7] = 0;
    i32x4 g2 = {};                                     // 2D tensor: groups 2/3 unused
    i32x4 g3 = {};
#if defined(__clang_major__) && (__clang_major__ >= 23)
    i32x8 g4 = {};
    __builtin_amdgcn_tensor_load_to_lds(g0, g1, g2, g3, g4, 0);
#else
    __builtin_amdgcn_tensor_load_to_lds(g0, g1, g2, g3, 0);
#endif
    __builtin_amdgcn_s_wait_tensorcnt(0);
  }
#endif

  const int mA   = lane & 15;
  const int koff = (lane >> 4) << 3;
  const unsigned short* arow = sb_prev + (size_t)(m0 + mA) * H_ + koff;

  v8f acc0 = {}, acc1 = {}, acc2 = {}, acc3 = {};
  v16bf a = {};
  if (!first) a = load_a_bf(arow);          // overlaps the Wh DMA

#if HAVE_TDM
  __syncthreads();                          // LDS slab visible to all 8 waves
#endif

  if (!first) {
#if HAVE_TDM
    const unsigned* bt = &lds_wh[0];
    const int bstride = 1024;
#else
    const unsigned* bt = Wsw + (size_t)(n0 >> 4) * 256;
    const int bstride = 32 * 256;
#endif
    v16bf b0v = load_b_sw(bt,       lane);
    v16bf b1v = load_b_sw(bt + 256, lane);
    v16bf b2v = load_b_sw(bt + 512, lane);
    v16bf b3v = load_b_sw(bt + 768, lane);
#pragma unroll
    for (int kk = 0; kk < 16; ++kk) {
      v16bf ca = a, c0 = b0v, c1 = b1v, c2 = b2v, c3 = b3v;
      if (kk < 15) {
        bt += bstride;
        a   = load_a_bf(arow + (kk + 1) * 32);
        b0v = load_b_sw(bt,       lane);
        b1v = load_b_sw(bt + 256, lane);
        b2v = load_b_sw(bt + 512, lane);
        b3v = load_b_sw(bt + 768, lane);
      }
      acc0 = WMMA_BF16(ca, c0, acc0);
      acc1 = WMMA_BF16(ca, c1, acc1);
      acc2 = WMMA_BF16(ca, c2, acc2);
      acc3 = WMMA_BF16(ca, c3, acc3);
    }
  }

  const int nc = lane & 15;
  const int mc = (lane >> 4) << 3;
#pragma unroll
  for (int i = 0; i < 8; ++i) {
    size_t ob = (size_t)(m0 + mc + i) * H_ + n0 + nc;
    float r0 = FAST_TANH(xp[ob]      + acc0[i]);
    float r1 = FAST_TANH(xp[ob + 16] + acc1[i]);
    float r2 = FAST_TANH(xp[ob + 32] + acc2[i]);
    float r3 = FAST_TANH(xp[ob + 48] + acc3[i]);
    xp[ob]      = r0;  xp[ob + 16] = r1;  xp[ob + 32] = r2;  xp[ob + 48] = r3;
    sb_next[ob]      = bf16_1(r0);        // bf16 state copy for the next step
    sb_next[ob + 16] = bf16_1(r1);
    sb_next[ob + 32] = bf16_1(r2);
    sb_next[ob + 48] = bf16_1(r3);
  }
}

// ---------------------------------------------------------------------------
// Host: swizzle weights, big GEMM into d_out, then 1024 stream-ordered steps
// with a bf16 state ping-pong in workspace.
// Workspace: 512K (Wx bf16) + 512K (Wh bf16) + 2 x 128K (bf16 state) = 1.25 MB.
// ---------------------------------------------------------------------------
extern "C" void kernel_launch(void* const* d_in, const int* in_sizes, int n_in,
                              void* d_out, int out_size, void* d_ws, size_t ws_size,
                              hipStream_t stream) {
  (void)in_sizes; (void)n_in; (void)out_size; (void)ws_size;
  const float* X    = (const float*)d_in[0];   // [B,T,H] f32
  // d_in[1] = initial state (all zeros) — folded into the t==0 step
  const float* Wx   = (const float*)d_in[2];   // [H,H] f32
  const float* Wh   = (const float*)d_in[3];   // [H,H] f32
  const float* bias = (const float*)d_in[4];   // [H]   f32

  float* out = (float*)d_out;                  // [T,B,H] f32

  unsigned* wx_sw = (unsigned*)d_ws;
  unsigned* wh_sw = wx_sw + 131072;            // +512 KB
  unsigned short* sb0 = (unsigned short*)(wh_sw + 131072);  // +512 KB
  unsigned short* sb1 = sb0 + (size_t)B_ * H_;              // +128 KB

  swizzle_w_bf16<<<64, 256, 0, stream>>>(Wx, wx_sw);
  swizzle_w_bf16<<<64, 256, 0, stream>>>(Wh, wh_sw);

  gemm_x_wx<<<8192, 256, 0, stream>>>(X, wx_sw, bias, out);

  const size_t BH = (size_t)B_ * H_;
  for (int t = 0; t < T_; ++t) {
    float* xp_t = out + (size_t)t * BH;
    const unsigned short* prev = (t & 1) ? sb0 : sb1;
    unsigned short*       next = (t & 1) ? sb1 : sb0;
    rnn_step<<<8, 256, 0, stream>>>(xp_t, prev, next, wh_sw, t == 0 ? 1 : 0);
  }
}
